// MambaBlock_80444737454249
// MI455X (gfx1250) — matrix operand order repaired and checked by reading, not compiled, not run on hardware
//
#include <hip/hip_runtime.h>
#include <hip/hip_bf16.h>
#include <cmath>
#include <cstdint>

#define D_MODEL 1024
#define D_STATE 16
#define D_CONVW 4
#define D_INNER 2048
#define DT_RANK 64
#define BB 2
#define LLEN 2048
#define MROWS (BB * LLEN)   // 4096
#define XPROJ_N (DT_RANK + 2 * D_STATE) // 96
#define LN_EPS 1e-5f

typedef __bf16 bf16;
typedef __attribute__((ext_vector_type(16))) __bf16 v16bf;
typedef __attribute__((ext_vector_type(8)))  float  v8f;

static __device__ __forceinline__ bf16 f2bf(float f) {
  unsigned u = __builtin_bit_cast(unsigned, f);
  unsigned r = (u + 0x7FFFu + ((u >> 16) & 1u)) >> 16;   // RNE
  unsigned short s = (unsigned short)r;
  return __builtin_bit_cast(bf16, s);
}
static __device__ __forceinline__ float bf2f(bf16 b) {
  unsigned short s = __builtin_bit_cast(unsigned short, b);
  unsigned u = ((unsigned)s) << 16;
  return __builtin_bit_cast(float, u);
}

// CDNA5 async global->LDS DMA (ASYNCcnt-tracked). offset: applies to both
// the LDS and global addresses (ISA 15.18), so one addr pair covers 2 chunks.
template <int OFF>
static __device__ __forceinline__ void async_ld_b128(unsigned lds_off, const void* g) {
  asm volatile("global_load_async_to_lds_b128 %0, %1, off offset:%2"
               :: "v"(lds_off),
                  "v"((unsigned long long)(uintptr_t)g),
                  "n"(OFF)
               : "memory");
}
static __device__ __forceinline__ void wait_async0() {
#if __has_builtin(__builtin_amdgcn_s_wait_asynccnt)
  __builtin_amdgcn_s_wait_asynccnt(0);
#else
  asm volatile("s_wait_asynccnt 0x0" ::: "memory");
#endif
}
static __device__ __forceinline__ unsigned lds_off32(const void* p) {
  // generic shared pointer: low 32 bits == LDS byte offset
  return (unsigned)(uintptr_t)p;
}

// ---------------------------------------------------------------- converts
__global__ void cvt_f32_bf16(const float* __restrict__ src, bf16* __restrict__ dst, int n) {
  int i = blockIdx.x * blockDim.x + threadIdx.x;
  if (i < n) dst[i] = f2bf(src[i]);
}

// ---------------------------------------------------------------- WMMA GEMM
// C[M,N] = A[M,K] (bf16, row-major) x W[N,K]^T (bf16, row-major), fp32 accum.
// Double-buffered LDS, async global->LDS staging pipelined against WMMA.
// EPI 0: Cf = v (fp32)
// EPI 1: Cb = bf16(v)
// EPI 2: Cf = softplus(v + bias[n])
// EPI 3: Cf = v (ld N), and for n < DT_RANK also Cb = bf16(v) (ld DT_RANK)
template <int EPI>
__global__ __launch_bounds__(256) void gemm_bf16_wmma(
    const bf16* __restrict__ A, const bf16* __restrict__ W,
    float* __restrict__ Cf, bf16* __restrict__ Cb,
    const float* __restrict__ bias, int M, int N, int K)
{
  constexpr int BM = 128, BN = 64, BK = 32, PAD = 8;
  __shared__ __align__(16) bf16 sA[2][BM][BK + PAD];
  __shared__ __align__(16) bf16 sB[2][BN][BK + PAD];

  const int tid  = threadIdx.x;
  const int wave = tid >> 5;     // 8 waves (wave32)
  const int lane = tid & 31;
  const int half = lane >> 4;
  const int l16  = lane & 15;
  const int wm   = wave >> 1;    // 0..3  -> 32-row strips
  const int wn   = wave & 1;     // 0..1  -> 32-col strips
  const int m0   = blockIdx.y * BM;
  const int n0   = blockIdx.x * BN;

  // staging assignment: A 32B/thread (2 async b128), B 16B/thread (1 async b128)
  const int arow = tid >> 1, acb = (tid & 1) * 16;
  const int brow = tid >> 2, bcb = (tid & 3) * 8;
  int bn = n0 + brow; if (bn > N - 1) bn = N - 1;   // clamp OOB rows (cols >= N never stored)

  const bf16* gA = &A[(size_t)(m0 + arow) * K + acb];
  const bf16* gB = &W[(size_t)bn * K + bcb];

  v8f acc[2][2] = {};
  const int nsteps = K / BK;

  // prologue: stage tile 0 into buffer 0
  {
    unsigned la = lds_off32(&sA[0][arow][acb]);
    unsigned lb = lds_off32(&sB[0][brow][bcb]);
    async_ld_b128<0>(la, gA);
    async_ld_b128<16>(la, gA);
    async_ld_b128<0>(lb, gB);
  }

  for (int i = 0; i < nsteps; i++) {
    wait_async0();        // our async writes for buf[i&1] have landed
    __syncthreads();      // everyone's landed; everyone's reads of buf[(i+1)&1] done
    if (i + 1 < nsteps) { // stage next tile; overlaps with compute below
      const int nb = (i + 1) & 1;
      const int kn = (i + 1) * BK;
      unsigned la = lds_off32(&sA[nb][arow][acb]);
      unsigned lb = lds_off32(&sB[nb][brow][bcb]);
      async_ld_b128<0>(la, gA + kn);
      async_ld_b128<16>(la, gA + kn);
      async_ld_b128<0>(lb, gB + kn);
    }
    const int cur = i & 1;

    // A fragments: lanes 0-15 carry K{0..15}, lanes 16-31 K{16..31}
    v16bf af[2], bf[2];
#pragma unroll
    for (int mi = 0; mi < 2; mi++) {
      const bf16* p0 = &sA[cur][wm * 32 + mi * 16 + l16][half * 16];
      const bf16* p1 = &sA[cur][wm * 32 + mi * 16 + l16][half * 16 + 8];
#pragma unroll
      for (int j = 0; j < 8; j++) { af[mi][j] = p0[j]; af[mi][8 + j] = p1[j]; }
    }
    // B fragments: lane group selects K half, column = l16
#pragma unroll
    for (int ni = 0; ni < 2; ni++) {
      const bf16* p = &sB[cur][wn * 32 + ni * 16 + l16][half * 16];
#pragma unroll
      for (int j = 0; j < 16; j++) bf[ni][j] = p[j];
    }

#pragma unroll
    for (int mi = 0; mi < 2; mi++)
#pragma unroll
      for (int ni = 0; ni < 2; ni++)
        acc[mi][ni] = __builtin_amdgcn_wmma_f32_16x16x32_bf16(
            false, af[mi], false, bf[ni], (short)0, acc[mi][ni], false, false);
  }

  // epilogue: C/D layout -> row = r + 8*half, col = l16 within each 16x16 tile
#pragma unroll
  for (int mi = 0; mi < 2; mi++) {
#pragma unroll
    for (int ni = 0; ni < 2; ni++) {
      const int colg = n0 + wn * 32 + ni * 16 + l16;
      if (colg >= N) continue;
#pragma unroll
      for (int r = 0; r < 8; r++) {
        const int rowg = m0 + wm * 32 + mi * 16 + r + 8 * half;
        const float v = acc[mi][ni][r];
        if (EPI == 0) {
          Cf[(size_t)rowg * N + colg] = v;
        } else if (EPI == 1) {
          Cb[(size_t)rowg * N + colg] = f2bf(v);
        } else if (EPI == 2) {
          float t = v + bias[colg];
          Cf[(size_t)rowg * N + colg] = (t > 20.f) ? t : log1pf(__expf(t));
        } else {
          Cf[(size_t)rowg * N + colg] = v;
          if (colg < DT_RANK) Cb[(size_t)rowg * DT_RANK + colg] = f2bf(v);
        }
      }
    }
  }
}

// ------------------------------------------------- causal depthwise conv + SiLU
__global__ __launch_bounds__(256) void conv_silu(
    const bf16* __restrict__ xzb, const float* __restrict__ Wc,
    const float* __restrict__ bc, bf16* __restrict__ ub)
{
  int idx = blockIdx.x * 256 + threadIdx.x;
  if (idx >= MROWS * D_INNER) return;
  int e = idx & (D_INNER - 1);
  int row = idx >> 11;            // D_INNER = 2048
  int l = row & (LLEN - 1);
  float acc = bc[e];
#pragma unroll
  for (int j = 0; j < D_CONVW; j++) {
    int lt = l - (D_CONVW - 1) + j;
    if (lt >= 0)
      acc += bf2f(xzb[(size_t)(row - (D_CONVW - 1) + j) * (2 * D_INNER) + e]) *
             Wc[e * D_CONVW + j];
  }
  float s = acc / (1.f + __expf(-acc));
  ub[(size_t)row * D_INNER + e] = f2bf(s);
}

// ---------------------------------------- selective scan + D-skip + SiLU gate
#define TCHUNK 8
__global__ __launch_bounds__(256) void scan_gate(
    const float* __restrict__ xdb, const float* __restrict__ delta,
    const bf16* __restrict__ ub, const bf16* __restrict__ xzb,
    const float* __restrict__ A_log, const float* __restrict__ D_skip,
    bf16* __restrict__ ygb)
{
  const int e = blockIdx.x * 256 + threadIdx.x;   // channel
  const int b = blockIdx.y;
  float Arow[D_STATE], h[D_STATE];
#pragma unroll
  for (int n = 0; n < D_STATE; n++) {
    Arow[n] = -__expf(A_log[(size_t)e * D_STATE + n]);
    h[n] = 0.f;
  }
  const float Dv = D_skip[e];
  __shared__ float sBC[TCHUNK][2 * D_STATE];

  for (int t0 = 0; t0 < LLEN; t0 += TCHUNK) {
    {
      const int tt = threadIdx.x >> 5;   // 8 timesteps
      const int j  = threadIdx.x & 31;   // 32 B/C values
      sBC[tt][j] = xdb[((size_t)b * LLEN + t0 + tt) * XPROJ_N + DT_RANK + j];
    }
    __syncthreads();
#pragma unroll
    for (int tt = 0; tt < TCHUNK; tt++) {
      const size_t row = (size_t)b * LLEN + t0 + tt;
      const float d = delta[row * D_INNER + e];
      const float uval = bf2f(ub[row * D_INNER + e]);
      const float du = d * uval;
      float y = 0.f;
#pragma unroll
      for (int n = 0; n < D_STATE; n++) {
        float hn = h[n] * __expf(d * Arow[n]) + du * sBC[tt][n];
        h[n] = hn;
        y += hn * sBC[tt][D_STATE + n];
      }
      const float zv = bf2f(xzb[row * (2 * D_INNER) + D_INNER + e]);
      const float g = zv / (1.f + __expf(-zv));
      ygb[row * D_INNER + e] = f2bf((y + uval * Dv) * g);
    }
    __syncthreads();
  }
}

// -------------------------------------------------- residual + LayerNorm
__global__ __launch_bounds__(256) void residual_layernorm(
    const float* __restrict__ mo, const float* __restrict__ x,
    const float* __restrict__ gamma, const float* __restrict__ beta,
    float* __restrict__ out)
{
  const int row = blockIdx.x;
  const int tid = threadIdx.x;
  float v[D_MODEL / 256];
  float s = 0.f, ss = 0.f;
#pragma unroll
  for (int i = 0; i < D_MODEL / 256; i++) {
    int c = tid + i * 256;
    float t = mo[(size_t)row * D_MODEL + c] + x[(size_t)row * D_MODEL + c];
    v[i] = t; s += t; ss += t * t;
  }
  __shared__ float rs[256], rss[256];
  rs[tid] = s; rss[tid] = ss;
  __syncthreads();
  for (int off = 128; off > 0; off >>= 1) {
    if (tid < off) { rs[tid] += rs[tid + off]; rss[tid] += rss[tid + off]; }
    __syncthreads();
  }
  const float mu = rs[0] * (1.f / D_MODEL);
  const float var = rss[0] * (1.f / D_MODEL) - mu * mu;
  const float inv = rsqrtf(var + LN_EPS);
#pragma unroll
  for (int i = 0; i < D_MODEL / 256; i++) {
    int c = tid + i * 256;
    out[(size_t)row * D_MODEL + c] = (v[i] - mu) * inv * gamma[c] + beta[c];
  }
}

// ----------------------------------------------------------------- launch
extern "C" void kernel_launch(void* const* d_in, const int* in_sizes, int n_in,
                              void* d_out, int out_size, void* d_ws, size_t ws_size,
                              hipStream_t stream) {
  const float* x      = (const float*)d_in[0];
  const float* W_in   = (const float*)d_in[1];
  const float* W_conv = (const float*)d_in[2];
  const float* b_conv = (const float*)d_in[3];
  const float* W_xprj = (const float*)d_in[4];
  const float* W_dt   = (const float*)d_in[5];
  const float* b_dt   = (const float*)d_in[6];
  const float* A_log  = (const float*)d_in[7];
  const float* D_skip = (const float*)d_in[8];
  const float* W_out  = (const float*)d_in[9];
  const float* gamma  = (const float*)d_in[10];
  const float* beta   = (const float*)d_in[11];
  float* out = (float*)d_out;
  (void)in_sizes; (void)n_in; (void)out_size; (void)ws_size;

  char* ws = (char*)d_ws;
  size_t off = 0;
  auto alloc = [&](size_t bytes) -> char* {
    char* p = ws + off;
    off = (off + bytes + 255) & ~(size_t)255;
    return p;
  };
  bf16*  xb    = (bf16*) alloc((size_t)MROWS * D_MODEL * 2);
  bf16*  winb  = (bf16*) alloc((size_t)2 * D_INNER * D_MODEL * 2);
  bf16*  wxpb  = (bf16*) alloc((size_t)XPROJ_N * D_INNER * 2);
  bf16*  wdtb  = (bf16*) alloc((size_t)D_INNER * DT_RANK * 2);
  bf16*  woutb = (bf16*) alloc((size_t)D_MODEL * D_INNER * 2);
  bf16*  xzb   = (bf16*) alloc((size_t)MROWS * 2 * D_INNER * 2);
  bf16*  ub    = (bf16*) alloc((size_t)MROWS * D_INNER * 2);
  float* xdb   = (float*)alloc((size_t)MROWS * XPROJ_N * 4);
  bf16*  dtb   = (bf16*) alloc((size_t)MROWS * DT_RANK * 2);
  float* delta = (float*)alloc((size_t)MROWS * D_INNER * 4);
  bf16*  ygb   = (bf16*) alloc((size_t)MROWS * D_INNER * 2);
  float* mo    = (float*)alloc((size_t)MROWS * D_MODEL * 4);

  dim3 blk(256);
  auto cvt = [&](const float* s, bf16* d, int n) {
    cvt_f32_bf16<<<(n + 255) / 256, blk, 0, stream>>>(s, d, n);
  };
  cvt(x,      xb,    MROWS * D_MODEL);
  cvt(W_in,   winb,  2 * D_INNER * D_MODEL);
  cvt(W_xprj, wxpb,  XPROJ_N * D_INNER);
  cvt(W_dt,   wdtb,  D_INNER * DT_RANK);
  cvt(W_out,  woutb, D_MODEL * D_INNER);

  // in_proj: [4096,1024] x [4096,1024]^T -> xz bf16 [4096,4096]
  gemm_bf16_wmma<1><<<dim3(2 * D_INNER / 64, MROWS / 128), blk, 0, stream>>>(
      xb, winb, nullptr, xzb, nullptr, MROWS, 2 * D_INNER, D_MODEL);

  // causal depthwise conv + SiLU -> u bf16
  conv_silu<<<(MROWS * D_INNER + 255) / 256, blk, 0, stream>>>(xzb, W_conv, b_conv, ub);

  // x_proj: [4096,2048] x [96,2048]^T -> xdb fp32 (+ dt slice bf16)
  gemm_bf16_wmma<3><<<dim3((XPROJ_N + 63) / 64, MROWS / 128), blk, 0, stream>>>(
      ub, wxpb, xdb, dtb, nullptr, MROWS, XPROJ_N, D_INNER);

  // dt_proj: [4096,64] x [2048,64]^T + bias, softplus -> delta fp32
  gemm_bf16_wmma<2><<<dim3(D_INNER / 64, MROWS / 128), blk, 0, stream>>>(
      dtb, wdtb, delta, nullptr, b_dt, MROWS, D_INNER, DT_RANK);

  // selective scan, fused D-skip + SiLU gate -> yg bf16
  scan_gate<<<dim3(D_INNER / 256, BB), blk, 0, stream>>>(
      xdb, delta, ub, xzb, A_log, D_skip, ygb);

  // out_proj: [4096,2048] x [1024,2048]^T -> mo fp32
  gemm_bf16_wmma<0><<<dim3(D_MODEL / 64, MROWS / 128), blk, 0, stream>>>(
      ygb, woutb, mo, nullptr, nullptr, MROWS, D_MODEL, D_INNER);

  // residual + LayerNorm -> out fp32
  residual_layernorm<<<MROWS, blk, 0, stream>>>(mo, x, gamma, beta, out);
}
